// HeteroGAT_30322469109983
// MI455X (gfx1250) — matrix operand order repaired and checked by reading, not compiled, not run on hardware
//
#include <hip/hip_runtime.h>
#include <hip/hip_bf16.h>
#include <stdint.h>

// Problem constants
#define NNODES  16384
#define IN_DIM  128
#define HID     64
#define NHEAD   4
#define OUT_DIM 349
#define OUT_PAD 352
#define NEDGE   262144
#define TPB     256

typedef __bf16 v16bf __attribute__((ext_vector_type(16)));
typedef float  v8f   __attribute__((ext_vector_type(8)));
typedef unsigned int u32x4 __attribute__((ext_vector_type(4)));

// ---------- helpers ----------
__device__ __forceinline__ unsigned short f2bf(float f) {
  unsigned int u = __float_as_uint(f);
  unsigned int r = u + 0x7FFFu + ((u >> 16) & 1u);   // RNE
  return (unsigned short)(r >> 16);
}
// order-preserving float<->uint encode for atomic segment-max
__device__ __forceinline__ unsigned int fenc(float f) {
  unsigned int b = __float_as_uint(f);
  return (b & 0x80000000u) ? ~b : (b | 0x80000000u);
}
__device__ __forceinline__ float fdec(unsigned int e) {
  unsigned int b = (e & 0x80000000u) ? (e & 0x7FFFFFFFu) : ~e;
  return __uint_as_float(b);
}

// ---------- elementwise prep kernels ----------
__global__ void k_scale_bf16(const float* __restrict__ x, const float* __restrict__ emb,
                             unsigned short* __restrict__ o, int total, int C) {
  int i = blockIdx.x * blockDim.x + threadIdx.x;
  if (i < total) o[i] = f2bf(x[i] * emb[i % C]);
}
__global__ void k_conv_bf16(const float* __restrict__ x, unsigned short* __restrict__ o, int total) {
  int i = blockIdx.x * blockDim.x + threadIdx.x;
  if (i < total) o[i] = f2bf(x[i]);
}
__global__ void k_relu_bf16(const float* __restrict__ x, unsigned short* __restrict__ o, int total) {
  int i = blockIdx.x * blockDim.x + threadIdx.x;
  if (i < total) o[i] = f2bf(fmaxf(x[i], 0.0f));
}
__global__ void k_pad_linW(const float* __restrict__ w, unsigned short* __restrict__ o) {
  int i = blockIdx.x * blockDim.x + threadIdx.x;
  if (i >= IN_DIM * OUT_PAD) return;
  int k = i / OUT_PAD, c = i % OUT_PAD;
  o[i] = (c < OUT_DIM) ? f2bf(w[k * OUT_DIM + c]) : (unsigned short)0;
}

// ---------- bf16 WMMA GEMM:  C(M x ncols, f32) = A(M x K, bf16 rm) * B(K x ncols_pad, bf16 rm) [+bias]
// B is staged into LDS with async-to-LDS b128 copies (ASYNCcnt), transposed once in LDS so every
// B fragment is two ds_load_b128. One wave owns a 16-row strip; 8 waves/block; M = gridDim.x*128.
__global__ __launch_bounds__(256) void k_gemm_bf16(
    const unsigned short* __restrict__ A, const unsigned short* __restrict__ B,
    const float* __restrict__ bias, float* __restrict__ C,
    int K, int Kpad, int ncols_pad, int ncols_real, int lda, int ldc)
{
  extern __shared__ unsigned short smem[];
  unsigned short* shBt  = smem;                                // [ncols_pad * Kpad] transposed
  unsigned short* shRaw = smem + (size_t)ncols_pad * Kpad;     // [K * ncols_pad]    raw copy

  int tid  = threadIdx.x;
  int lane = tid & 31;
  int wave = tid >> 5;

  // ---- phase 1: async copy raw B (contiguous) into LDS; sizes are exact multiples of 256*16B ----
  {
    int totalBytes = K * ncols_pad * 2;
    unsigned ldsBase = (unsigned)(size_t)(void*)shRaw;         // low 32 bits = LDS byte offset
    for (int byteOff = tid * 16; byteOff < totalBytes; byteOff += TPB * 16) {
      unsigned ldsOff = ldsBase + (unsigned)byteOff;
      const void* gptr = (const char*)B + byteOff;
      asm volatile("global_load_async_to_lds_b128 %0, %1, off"
                   :: "v"(ldsOff), "v"(gptr) : "memory");
    }
    asm volatile("s_wait_asynccnt 0" ::: "memory");
    __syncthreads();
  }
  // ---- phase 2: transpose in LDS: shBt[n*Kpad + k] = shRaw[k*ncols_pad + n] ----
  for (int i = tid; i < K * ncols_pad; i += TPB) {
    int k = i / ncols_pad, n = i % ncols_pad;
    shBt[n * Kpad + k] = shRaw[i];
  }
  __syncthreads();

  int row0 = (blockIdx.x * 8 + wave) * 16;
  int rsub = lane >> 4;       // lane-half: 0 or 1
  int rlo  = lane & 15;
  int ntiles = ncols_pad >> 4;

  for (int t = 0; t < ntiles; ++t) {
    int col0 = t * 16;
    v8f acc = {};
    for (int k0 = 0; k0 < K; k0 += 32) {
      // A fragment: 16x32 bf16. element (j,p): k = k0 + 8*rsub + 16*(j/4) + 2*(j%4) + p
      union { v16bf v; unsigned int u[8]; } fa;
      const unsigned short* ap = A + (size_t)(row0 + rlo) * lda + k0 + rsub * 8;
      #pragma unroll
      for (int j = 0; j < 8; ++j) {
        int k = ((j >> 2) << 4) + ((j & 3) << 1);
        fa.u[j] = *(const unsigned int*)(ap + k);   // bf16 pair (k, k+1)
      }
      // B fragment from LDS: lane holds col = col0+rlo, k = k0 + 8*rsub + i (i<8), k0 + 16 + 8*rsub + (i-8) (i>=8)
      union { v16bf v; u32x4 q[2]; } fb;
      const unsigned short* bp = shBt + (size_t)(col0 + rlo) * Kpad + k0 + rsub * 8;
      fb.q[0] = *(const u32x4*)(bp);        // ds_load_b128
      fb.q[1] = *(const u32x4*)(bp + 16);   // ds_load_b128

      acc = __builtin_amdgcn_wmma_f32_16x16x32_bf16(false, fa.v, false, fb.v,
                                                    (short)0, acc, false, false);
    }
    int col = col0 + rlo;
    if (col < ncols_real) {
      float bv = bias ? bias[col] : 0.0f;
      int rbase = row0 + rsub * 8;
      #pragma unroll
      for (int r = 0; r < 8; ++r)
        C[(size_t)(rbase + r) * ldc + col] = acc[r] + bv;
    }
  }
}

// ---------- attention logits: e[n][h] = sum_d f[n][h*16+d] * a[h][d] ----------
__global__ void k_logits(const float* __restrict__ f, const float* __restrict__ avec,
                         float* __restrict__ e, int total) {
  int idx = blockIdx.x * blockDim.x + threadIdx.x;
  if (idx >= total) return;
  int n = idx >> 2, h = idx & 3;
  const float* fp = f + (size_t)n * HID + h * 16;
  const float* ap = avec + h * 16;
  float s = 0.0f;
  #pragma unroll
  for (int d = 0; d < 16; ++d) s += fp[d] * ap[d];
  e[idx] = s;
}

// ---------- edge pass 1: segment max of leaky_relu(el[src]+er[dst]) via ordered-uint atomicMax ----------
__global__ void k_edge_max(const int* __restrict__ src, const int* __restrict__ dst,
                           const float* __restrict__ el, const float* __restrict__ er,
                           unsigned int* __restrict__ menc, int ne) {
  int idx = blockIdx.x * blockDim.x + threadIdx.x;
  if (idx >= ne * 4) return;
  int e = idx >> 2, h = idx & 3;
  int s = src[e], d = dst[e];
  float v = el[s * 4 + h] + er[d * 4 + h];
  v = v > 0.0f ? v : 0.2f * v;
  atomicMax(&menc[d * 4 + h], fenc(v));
}

// decode max in place; nodes with no in-edges (enc still 0) -> m = 0 (matches jnp.where(isfinite(m),m,0))
__global__ void k_fin_m(unsigned int* __restrict__ m, int total) {
  int i = blockIdx.x * blockDim.x + threadIdx.x;
  if (i >= total) return;
  unsigned int u = m[i];
  float v = (u <= 0x007FFFFFu) ? 0.0f : fdec(u);   // 0x007FFFFF == enc(-inf)
  ((float*)m)[i] = v;
}

// ---------- edge pass 2: acc[dst] += fs[src]*exp(e-m), den[dst] += exp(e-m)  (unnormalized softmax) ----------
__global__ void k_edge_acc(const int* __restrict__ src, const int* __restrict__ dst,
                           const float* __restrict__ el, const float* __restrict__ er,
                           const float* __restrict__ m, const float* __restrict__ fs,
                           float* __restrict__ acc, float* __restrict__ den, int ne) {
  int idx = blockIdx.x * blockDim.x + threadIdx.x;
  if (idx >= ne * 4) return;
  int e = idx >> 2, h = idx & 3;
  int s = src[e], d = dst[e];
  const float* fp = fs + (size_t)s * HID + h * 16;
  __builtin_prefetch(fp, 0, 1);                     // global_prefetch_b8 on gathered row
  float v = el[s * 4 + h] + er[d * 4 + h];
  v = v > 0.0f ? v : 0.2f * v;
  float ex = expf(v - m[d * 4 + h]);
  atomicAdd(&den[d * 4 + h], ex);
  float* ap = acc + (size_t)d * HID + h * 16;
  #pragma unroll
  for (int j = 0; j < 16; ++j) atomicAdd(&ap[j], fp[j] * ex);
}

// ---------- GAT finalize: out = acc/max(den,1e-9) + bias  (accumulate=1 adds into out) ----------
__global__ void k_gat_fin(const float* __restrict__ acc, const float* __restrict__ den,
                          const float* __restrict__ b, float* __restrict__ out,
                          int total, int accumulate) {
  int idx = blockIdx.x * blockDim.x + threadIdx.x;
  if (idx >= total) return;
  int n = idx >> 6, c = idx & 63, h = c >> 4;
  float v = acc[idx] / fmaxf(den[n * 4 + h], 1e-9f) + b[c];
  if (accumulate) out[idx] += v; else out[idx] = v;
}

// ---------- degenerate transformer (dim-1 LN => h2 = h1 + c1) + per-row L2 norms -> hf (bf16) ----------
__global__ void k_rownorm_hf(const float* __restrict__ h1, const float* __restrict__ attn_w,
                             const float* __restrict__ attn_b, unsigned short* __restrict__ hf, int N) {
  int n = blockIdx.x * blockDim.x + threadIdx.x;
  if (n >= N) return;
  float c1 = attn_b[2] * attn_w[3] + attn_b[3];   // exact collapse of the size-1-dim attention block
  const float* p = h1 + (size_t)n * HID;
  float s1 = 0.0f, s2 = 0.0f;
  #pragma unroll 8
  for (int c = 0; c < HID; ++c) { float a = p[c]; float b = a + c1; s1 += a * a; s2 += b * b; }
  float r1 = 1.0f / fmaxf(sqrtf(s1), 1e-12f);
  float r2 = 1.0f / fmaxf(sqrtf(s2), 1e-12f);
  unsigned short* q = hf + (size_t)n * (2 * HID);
  #pragma unroll 8
  for (int c = 0; c < HID; ++c) { q[c] = f2bf(p[c] * r1); q[HID + c] = f2bf((p[c] + c1) * r2); }
}

// ============================================================================
extern "C" void kernel_launch(void* const* d_in, const int* in_sizes, int n_in,
                              void* d_out, int out_size, void* d_ws, size_t ws_size,
                              hipStream_t stream) {
  (void)in_sizes; (void)n_in; (void)out_size; (void)ws_size;
  const float* x_author = (const float*)d_in[0];
  const float* x_paper  = (const float*)d_in[1];
  const float* ntype    = (const float*)d_in[2];
  const float* W0       = (const float*)d_in[3];
  const float* al0      = (const float*)d_in[4];
  const float* ar0      = (const float*)d_in[5];
  const float* b0       = (const float*)d_in[6];
  const float* W1       = (const float*)d_in[7];
  const float* al1      = (const float*)d_in[8];
  const float* ar1      = (const float*)d_in[9];
  const float* b1       = (const float*)d_in[10];
  const float* attn_w   = (const float*)d_in[12];
  const float* attn_b   = (const float*)d_in[13];
  const float* lin_W    = (const float*)d_in[19];
  const float* lin_b    = (const float*)d_in[20];
  const int* src_w = (const int*)d_in[21];
  const int* dst_w = (const int*)d_in[22];
  const int* src_c = (const int*)d_in[23];
  const int* dst_c = (const int*)d_in[24];
  const int* src_r = (const int*)d_in[25];
  const int* dst_r = (const int*)d_in[26];
  float* out = (float*)d_out;

  // workspace carve-up (everything fits easily in L2: ~66MB total)
  char* base = (char*)d_ws; size_t off = 0;
  auto alloc = [&](size_t bytes) -> void* {
    void* p = base + off; off = (off + bytes + 255) & ~(size_t)255; return p;
  };
  unsigned short* ha_bf  = (unsigned short*)alloc((size_t)NNODES * IN_DIM * 2);
  unsigned short* hp_bf  = (unsigned short*)alloc((size_t)NNODES * IN_DIM * 2);
  unsigned short* W0b    = (unsigned short*)alloc((size_t)3 * IN_DIM * HID * 2);
  unsigned short* W1b    = (unsigned short*)alloc((size_t)3 * HID * HID * 2);
  unsigned short* linWb  = (unsigned short*)alloc((size_t)IN_DIM * OUT_PAD * 2);
  float* fsW = (float*)alloc((size_t)NNODES * HID * 4);
  float* fdW = (float*)alloc((size_t)NNODES * HID * 4);
  float* fC  = (float*)alloc((size_t)NNODES * HID * 4);
  float* fsR = (float*)alloc((size_t)NNODES * HID * 4);
  float* fdR = (float*)alloc((size_t)NNODES * HID * 4);
  float* elW = (float*)alloc((size_t)NNODES * 4 * 4);
  float* erW = (float*)alloc((size_t)NNODES * 4 * 4);
  float* elC = (float*)alloc((size_t)NNODES * 4 * 4);
  float* erC = (float*)alloc((size_t)NNODES * 4 * 4);
  float* elR = (float*)alloc((size_t)NNODES * 4 * 4);
  float* erR = (float*)alloc((size_t)NNODES * 4 * 4);
  unsigned int* mW = (unsigned int*)alloc((size_t)NNODES * 4 * 4);
  unsigned int* mC = (unsigned int*)alloc((size_t)NNODES * 4 * 4);
  unsigned int* mR = (unsigned int*)alloc((size_t)NNODES * 4 * 4);
  float* denW = (float*)alloc((size_t)NNODES * 4 * 4);
  float* denC = (float*)alloc((size_t)NNODES * 4 * 4);
  float* denR = (float*)alloc((size_t)NNODES * 4 * 4);
  float* accW = (float*)alloc((size_t)NNODES * HID * 4);
  float* accC = (float*)alloc((size_t)NNODES * HID * 4);
  float* accR = (float*)alloc((size_t)NNODES * HID * 4);
  float* op   = (float*)alloc((size_t)NNODES * HID * 4);
  float* oa   = (float*)alloc((size_t)NNODES * HID * 4);
  unsigned short* ha1_bf = (unsigned short*)alloc((size_t)NNODES * HID * 2);
  unsigned short* hp1_bf = (unsigned short*)alloc((size_t)NNODES * HID * 2);
  float* h1 = (float*)alloc((size_t)NNODES * HID * 4);
  unsigned short* hf_bf = (unsigned short*)alloc((size_t)NNODES * 2 * HID * 2);

  const int N128 = NNODES * IN_DIM, N64 = NNODES * HID, N4 = NNODES * 4, E4 = NEDGE * 4;
  dim3 blk(TPB);
  dim3 gN128((N128 + TPB - 1) / TPB), gN64((N64 + TPB - 1) / TPB);
  dim3 gN4((N4 + TPB - 1) / TPB), gE4((E4 + TPB - 1) / TPB);
  dim3 gN((NNODES + TPB - 1) / TPB);
  dim3 gGemm(NNODES / (16 * 8));   // 8 waves/block, 16 rows/wave

  // dynamic LDS: transposed B [ncols_pad*Kpad] + raw B [K*ncols_pad], bf16
  const size_t shm_feat = (size_t)(HID * 136 + IN_DIM * HID) * 2;          // K=128,Kpad=136,N=64
  const size_t shm_l1   = (size_t)(HID * 72  + HID * HID) * 2;             // K=64, Kpad=72, N=64
  const size_t shm_fin  = (size_t)(OUT_PAD * 136 + IN_DIM * OUT_PAD) * 2;  // K=128,Kpad=136,N=352

  // ---- prep: scale by node-type embedding, convert weights to bf16 ----
  k_scale_bf16<<<gN128, blk, 0, stream>>>(x_author, ntype,          ha_bf, N128, IN_DIM);
  k_scale_bf16<<<gN128, blk, 0, stream>>>(x_paper,  ntype + IN_DIM, hp_bf, N128, IN_DIM);
  k_conv_bf16<<<dim3((3 * IN_DIM * HID + TPB - 1) / TPB), blk, 0, stream>>>(W0, W0b, 3 * IN_DIM * HID);
  k_conv_bf16<<<dim3((3 * HID * HID + TPB - 1) / TPB),    blk, 0, stream>>>(W1, W1b, 3 * HID * HID);
  k_pad_linW<<<dim3((IN_DIM * OUT_PAD + TPB - 1) / TPB),  blk, 0, stream>>>(lin_W, linWb);

  // ---- layer 0 feature GEMMs (WMMA bf16, B staged via async-to-LDS) ----
  k_gemm_bf16<<<gGemm, blk, shm_feat, stream>>>(ha_bf, W0b,                    nullptr, fsW, IN_DIM, 136, HID, HID, IN_DIM, HID);
  k_gemm_bf16<<<gGemm, blk, shm_feat, stream>>>(hp_bf, W0b,                    nullptr, fdW, IN_DIM, 136, HID, HID, IN_DIM, HID);
  k_gemm_bf16<<<gGemm, blk, shm_feat, stream>>>(hp_bf, W0b + 1 * IN_DIM * HID, nullptr, fC,  IN_DIM, 136, HID, HID, IN_DIM, HID);
  k_gemm_bf16<<<gGemm, blk, shm_feat, stream>>>(hp_bf, W0b + 2 * IN_DIM * HID, nullptr, fsR, IN_DIM, 136, HID, HID, IN_DIM, HID);
  k_gemm_bf16<<<gGemm, blk, shm_feat, stream>>>(ha_bf, W0b + 2 * IN_DIM * HID, nullptr, fdR, IN_DIM, 136, HID, HID, IN_DIM, HID);

  // ---- layer 0 logits ----
  k_logits<<<gN4, blk, 0, stream>>>(fsW, al0,       elW, N4);
  k_logits<<<gN4, blk, 0, stream>>>(fdW, ar0,       erW, N4);
  k_logits<<<gN4, blk, 0, stream>>>(fC,  al0 + 64,  elC, N4);
  k_logits<<<gN4, blk, 0, stream>>>(fC,  ar0 + 64,  erC, N4);
  k_logits<<<gN4, blk, 0, stream>>>(fsR, al0 + 128, elR, N4);
  k_logits<<<gN4, blk, 0, stream>>>(fdR, ar0 + 128, erR, N4);

  // ---- zero accumulators ----
  hipMemsetAsync(mW,   0, (size_t)N4 * 4, stream);
  hipMemsetAsync(mC,   0, (size_t)N4 * 4, stream);
  hipMemsetAsync(mR,   0, (size_t)N4 * 4, stream);
  hipMemsetAsync(denW, 0, (size_t)N4 * 4, stream);
  hipMemsetAsync(denC, 0, (size_t)N4 * 4, stream);
  hipMemsetAsync(denR, 0, (size_t)N4 * 4, stream);
  hipMemsetAsync(accW, 0, (size_t)N64 * 4, stream);
  hipMemsetAsync(accC, 0, (size_t)N64 * 4, stream);
  hipMemsetAsync(accR, 0, (size_t)N64 * 4, stream);

  // ---- layer 0 edge softmax + aggregation ----
  k_edge_max<<<gE4, blk, 0, stream>>>(src_w, dst_w, elW, erW, mW, NEDGE);
  k_edge_max<<<gE4, blk, 0, stream>>>(src_c, dst_c, elC, erC, mC, NEDGE);
  k_edge_max<<<gE4, blk, 0, stream>>>(src_r, dst_r, elR, erR, mR, NEDGE);
  k_fin_m<<<gN4, blk, 0, stream>>>(mW, N4);
  k_fin_m<<<gN4, blk, 0, stream>>>(mC, N4);
  k_fin_m<<<gN4, blk, 0, stream>>>(mR, N4);
  k_edge_acc<<<gE4, blk, 0, stream>>>(src_w, dst_w, elW, erW, (float*)mW, fsW, accW, denW, NEDGE);
  k_edge_acc<<<gE4, blk, 0, stream>>>(src_c, dst_c, elC, erC, (float*)mC, fC,  accC, denC, NEDGE);
  k_edge_acc<<<gE4, blk, 0, stream>>>(src_r, dst_r, elR, erR, (float*)mR, fsR, accR, denR, NEDGE);
  k_gat_fin<<<gN64, blk, 0, stream>>>(accW, denW, b0,       op, N64, 0);
  k_gat_fin<<<gN64, blk, 0, stream>>>(accC, denC, b0 + 64,  op, N64, 1);
  k_gat_fin<<<gN64, blk, 0, stream>>>(accR, denR, b0 + 128, oa, N64, 0);

  // ---- relu + bf16 for layer 1 ----
  k_relu_bf16<<<gN64, blk, 0, stream>>>(oa, ha1_bf, N64);
  k_relu_bf16<<<gN64, blk, 0, stream>>>(op, hp1_bf, N64);

  // ---- layer 1 (author output is dead code in reference -> only writes+cites GATs) ----
  k_gemm_bf16<<<gGemm, blk, shm_l1, stream>>>(ha1_bf, W1b,             nullptr, fsW, HID, 72, HID, HID, HID, HID);
  k_gemm_bf16<<<gGemm, blk, shm_l1, stream>>>(hp1_bf, W1b,             nullptr, fdW, HID, 72, HID, HID, HID, HID);
  k_gemm_bf16<<<gGemm, blk, shm_l1, stream>>>(hp1_bf, W1b + HID * HID, nullptr, fC,  HID, 72, HID, HID, HID, HID);
  k_logits<<<gN4, blk, 0, stream>>>(fsW, al1,      elW, N4);
  k_logits<<<gN4, blk, 0, stream>>>(fdW, ar1,      erW, N4);
  k_logits<<<gN4, blk, 0, stream>>>(fC,  al1 + 64, elC, N4);
  k_logits<<<gN4, blk, 0, stream>>>(fC,  ar1 + 64, erC, N4);
  hipMemsetAsync(mW,   0, (size_t)N4 * 4, stream);
  hipMemsetAsync(mC,   0, (size_t)N4 * 4, stream);
  hipMemsetAsync(denW, 0, (size_t)N4 * 4, stream);
  hipMemsetAsync(denC, 0, (size_t)N4 * 4, stream);
  hipMemsetAsync(accW, 0, (size_t)N64 * 4, stream);
  hipMemsetAsync(accC, 0, (size_t)N64 * 4, stream);
  k_edge_max<<<gE4, blk, 0, stream>>>(src_w, dst_w, elW, erW, mW, NEDGE);
  k_edge_max<<<gE4, blk, 0, stream>>>(src_c, dst_c, elC, erC, mC, NEDGE);
  k_fin_m<<<gN4, blk, 0, stream>>>(mW, N4);
  k_fin_m<<<gN4, blk, 0, stream>>>(mC, N4);
  k_edge_acc<<<gE4, blk, 0, stream>>>(src_w, dst_w, elW, erW, (float*)mW, fsW, accW, denW, NEDGE);
  k_edge_acc<<<gE4, blk, 0, stream>>>(src_c, dst_c, elC, erC, (float*)mC, fC,  accC, denC, NEDGE);
  k_gat_fin<<<gN64, blk, 0, stream>>>(accW, denW, b1,      h1, N64, 0);
  k_gat_fin<<<gN64, blk, 0, stream>>>(accC, denC, b1 + 64, h1, N64, 1);

  // ---- collapsed transformer + L2-normalized concat -> hf (bf16) ----
  k_rownorm_hf<<<gN, blk, 0, stream>>>(h1, attn_w, attn_b, hf_bf, NNODES);

  // ---- final linear (WMMA bf16, padded 349->352, guarded stores) ----
  k_gemm_bf16<<<gGemm, blk, shm_fin, stream>>>(hf_bf, linWb, lin_b, out,
                                               2 * HID, 136, OUT_PAD, OUT_DIM, 2 * HID, OUT_DIM);
}
